// GraphGCN_57870389346569
// MI455X (gfx1250) — matrix operand order repaired and checked by reading, not compiled, not run on hardware
//
#include <hip/hip_runtime.h>
#include <hip/hip_bf16.h>

// ---------------------------------------------------------------------------
// GCN forward on gfx1250 (MI455X): LN -> GEMM(bf16 WMMA) -> scatter -> LN+ReLU
//                                 -> GEMM -> scatter -> LN+ReLU+pool -> logits
// ---------------------------------------------------------------------------

typedef __attribute__((ext_vector_type(16))) __bf16 v16bf;
typedef __attribute__((ext_vector_type(8)))  float  v8f;

#define N_NODES 100000
#define N_EDGES 1600000
#define NF      128
#define NCLASS  16
#define NGRAPH  64
#define EPS_LN  1e-5f

// ---------------- small helpers ----------------

static __device__ __forceinline__ float wave_sum32(float v) {
    #pragma unroll
    for (int m = 16; m >= 1; m >>= 1) v += __shfl_xor(v, m, 32);
    return v;
}

static __device__ __forceinline__ v16bf load_bf16_frag(const __bf16* p0, const __bf16* p1) {
    union { uint4 u[2]; v16bf v; } r;
    r.u[0] = *reinterpret_cast<const uint4*>(p0);   // 8 bf16 = 16B (aligned)
    r.u[1] = *reinterpret_cast<const uint4*>(p1);
    return r.v;
}

// ---------------- kernels ----------------

__global__ void zero_f32(float* __restrict__ p, int n) {
    int i = blockIdx.x * blockDim.x + threadIdx.x;
    if (i < n) p[i] = 0.0f;
}

__global__ void convert_w_bf16(const float* __restrict__ W1, const float* __restrict__ W2,
                               __bf16* __restrict__ wbf) {
    int i = blockIdx.x * blockDim.x + threadIdx.x;
    if (i < NF * NF)          wbf[i] = (__bf16)W1[i];
    else if (i < 2 * NF * NF) wbf[i] = (__bf16)W2[i - NF * NF];
}

// deg[dst] += ew   (self-loop +1 folded into finalize)
__global__ void deg_kernel(const int* __restrict__ dst, const float* __restrict__ ew,
                           float* __restrict__ deg) {
    int e = blockIdx.x * blockDim.x + threadIdx.x;
    if (e < N_EDGES) unsafeAtomicAdd(&deg[dst[e]], ew[e]);
}

__global__ void finalize_dinv(float* __restrict__ deg_dinv) {
    int n = blockIdx.x * blockDim.x + threadIdx.x;
    if (n < N_NODES) {
        float d = deg_dinv[n] + 1.0f;      // + self-loop weight
        deg_dinv[n] = rsqrtf(d);           // d >= 1 always
    }
}

// LayerNorm (+optional ReLU) over 128 feats, one wave per node, bf16 output.
__global__ void ln_to_bf16(const float* __restrict__ in,
                           const float* __restrict__ g, const float* __restrict__ b,
                           __bf16* __restrict__ out, int do_relu) {
    int lane = threadIdx.x & 31;
    int nid  = blockIdx.x * (blockDim.x >> 5) + (threadIdx.x >> 5);
    if (nid >= N_NODES) return;
    const float* row = in + (size_t)nid * NF;
    float v[4];
    float s = 0.f;
    #pragma unroll
    for (int j = 0; j < 4; ++j) { v[j] = row[lane + 32 * j]; s += v[j]; }
    float mu = wave_sum32(s) * (1.0f / NF);
    float q = 0.f;
    #pragma unroll
    for (int j = 0; j < 4; ++j) { float d = v[j] - mu; q += d * d; }
    float rstd = rsqrtf(wave_sum32(q) * (1.0f / NF) + EPS_LN);
    __bf16* orow = out + (size_t)nid * NF;
    #pragma unroll
    for (int j = 0; j < 4; ++j) {
        int f = lane + 32 * j;
        float y = (v[j] - mu) * rstd * g[f] + b[f];
        if (do_relu) y = fmaxf(y, 0.f);
        orow[f] = (__bf16)y;
    }
}

// LayerNorm + ReLU + global_add_pool (atomic into pooled[graph][feat]).
__global__ void ln_relu_pool(const float* __restrict__ in,
                             const float* __restrict__ g, const float* __restrict__ b,
                             const int* __restrict__ batch,
                             float* __restrict__ pooled) {
    int lane = threadIdx.x & 31;
    int nid  = blockIdx.x * (blockDim.x >> 5) + (threadIdx.x >> 5);
    if (nid >= N_NODES) return;
    const float* row = in + (size_t)nid * NF;
    float v[4];
    float s = 0.f;
    #pragma unroll
    for (int j = 0; j < 4; ++j) { v[j] = row[lane + 32 * j]; s += v[j]; }
    float mu = wave_sum32(s) * (1.0f / NF);
    float q = 0.f;
    #pragma unroll
    for (int j = 0; j < 4; ++j) { float d = v[j] - mu; q += d * d; }
    float rstd = rsqrtf(wave_sum32(q) * (1.0f / NF) + EPS_LN);
    int gid = batch[nid];
    float* prow = pooled + (size_t)gid * NF;
    #pragma unroll
    for (int j = 0; j < 4; ++j) {
        int f = lane + 32 * j;
        float y = fmaxf((v[j] - mu) * rstd * g[f] + b[f], 0.f);
        unsafeAtomicAdd(&prow[f], y);
    }
}

// C[N,128] = A[N,128](bf16) @ W[128,128](bf16, row-major [out,in])^T, fp32 acc.
// Block = 256 thr = 8 waves; wave w -> 16-col tile; block -> 16-row node tile.
__global__ void gemm_bf16_wmma(const __bf16* __restrict__ A, const __bf16* __restrict__ W,
                               float* __restrict__ C) {
    int wave = threadIdx.x >> 5;
    int lane = threadIdx.x & 31;
    int lr = lane & 15;            // row (A) / col (B) within tile
    int lh = lane >> 4;            // lane-half selector
    int row0 = blockIdx.x * 16;    // N_NODES % 16 == 0
    int col0 = wave * 16;

    const __bf16* arow = A + (size_t)(row0 + lr) * NF + lh * 8;
    const __bf16* wrow = W + (size_t)(col0 + lr) * NF + lh * 8;

    v8f acc = {0.f, 0.f, 0.f, 0.f, 0.f, 0.f, 0.f, 0.f};
    #pragma unroll
    for (int kc = 0; kc < 4; ++kc) {
        // A 16x32 bf16 frag: elems 0-7 -> K = lh*8 + kc*32 .. ; elems 8-15 -> +16
        v16bf a = load_bf16_frag(arow + kc * 32, arow + kc * 32 + 16);
        // B 32x16 frag: col = col0+lr, elems 0-7 -> K = lh*8 + kc*32 .. ; elems 8-15 -> +16
        v16bf b = load_bf16_frag(wrow + kc * 32, wrow + kc * 32 + 16);
        acc = __builtin_amdgcn_wmma_f32_16x16x32_bf16(
            /*neg_a=*/false, a, /*neg_b=*/false, b,
            /*c_mod=*/(short)0, acc, /*reuse_a=*/false, /*reuse_b=*/false);
    }
    // D: elem j -> (m = row0 + j + 8*lh, n = col0 + lr)
    #pragma unroll
    for (int j = 0; j < 8; ++j)
        C[(size_t)(row0 + j + 8 * lh) * NF + col0 + lr] = acc[j];
}

// agg = hW * dinv^2 (self loop, norm = dinv[n]*1*dinv[n]) + bias  (full overwrite)
__global__ void init_agg(const float* __restrict__ hw, const float* __restrict__ dinv,
                         const float* __restrict__ bias, float* __restrict__ agg) {
    size_t idx = (size_t)blockIdx.x * blockDim.x + threadIdx.x;
    if (idx >= (size_t)N_NODES * NF) return;
    int n = (int)(idx >> 7);
    int f = (int)(idx & 127);
    float di = dinv[n];
    agg[idx] = hw[idx] * di * di + bias[f];
}

// agg[dst] += hW[src] * (dinv[src]*ew*dinv[dst]); thread per (edge, feature)
__global__ void edge_scatter(const int* __restrict__ src, const int* __restrict__ dst,
                             const float* __restrict__ ew, const float* __restrict__ dinv,
                             const float* __restrict__ hw, float* __restrict__ agg) {
    size_t idx = (size_t)blockIdx.x * blockDim.x + threadIdx.x;
    if (idx >= (size_t)N_EDGES * NF) return;
    int e = (int)(idx >> 7);
    int f = (int)(idx & 127);
    int s = src[e], d = dst[e];
    float nm = dinv[s] * ew[e] * dinv[d];
    unsafeAtomicAdd(&agg[(size_t)d * NF + f], hw[(size_t)s * NF + f] * nm);
}

// logits = pooled @ Wc^T + bc ; log_softmax over 16 classes. 1 block/graph, 32 lanes.
__global__ void classifier(const float* __restrict__ pooled, const float* __restrict__ Wc,
                           const float* __restrict__ bc, float* __restrict__ out) {
    int g = blockIdx.x;
    int lane = threadIdx.x;
    float logit = -INFINITY;
    if (lane < NCLASS) {
        float s = bc[lane];
        const float* p = pooled + (size_t)g * NF;
        const float* w = Wc + (size_t)lane * NF;
        #pragma unroll 4
        for (int k = 0; k < NF; ++k) s += p[k] * w[k];
        logit = s;
    }
    // reductions stay within the 16-lane group (xor masks <= 8)
    float m = logit;
    #pragma unroll
    for (int x = 8; x >= 1; x >>= 1) m = fmaxf(m, __shfl_xor(m, x, 32));
    float ex = (lane < NCLASS) ? __expf(logit - m) : 0.f;
    float ss = ex;
    #pragma unroll
    for (int x = 8; x >= 1; x >>= 1) ss += __shfl_xor(ss, x, 32);
    if (lane < NCLASS) out[(size_t)g * NCLASS + lane] = logit - m - __logf(ss);
}

// ---------------------------------------------------------------------------

static inline size_t align256(size_t x) { return (x + 255) & ~(size_t)255; }

extern "C" void kernel_launch(void* const* d_in, const int* in_sizes, int n_in,
                              void* d_out, int out_size, void* d_ws, size_t ws_size,
                              hipStream_t stream) {
    (void)in_sizes; (void)n_in; (void)out_size; (void)ws_size;

    const float* x    = (const float*)d_in[0];
    const int*   ei   = (const int*)  d_in[1];     // [2, E] flat
    const int*   batch= (const int*)  d_in[2];
    const float* ew   = (const float*)d_in[3];
    const float* ln0g = (const float*)d_in[4];
    const float* ln0b = (const float*)d_in[5];
    const float* W1   = (const float*)d_in[6];
    const float* b1   = (const float*)d_in[7];
    const float* ln1g = (const float*)d_in[8];
    const float* ln1b = (const float*)d_in[9];
    const float* W2   = (const float*)d_in[10];
    const float* b2   = (const float*)d_in[11];
    const float* ln2g = (const float*)d_in[12];
    const float* ln2b = (const float*)d_in[13];
    const float* Wc   = (const float*)d_in[14];
    const float* bc   = (const float*)d_in[15];
    const int* src = ei;
    const int* dst = ei + N_EDGES;
    float* out = (float*)d_out;

    // workspace layout
    char* ws = (char*)d_ws;
    size_t off = 0;
    float*  dinv   = (float*) (ws + off); off = align256(off + (size_t)N_NODES * 4);
    __bf16* hbf    = (__bf16*)(ws + off); off = align256(off + (size_t)N_NODES * NF * 2);
    float*  hw     = (float*) (ws + off); off = align256(off + (size_t)N_NODES * NF * 4);
    float*  agg    = (float*) (ws + off); off = align256(off + (size_t)N_NODES * NF * 4);
    __bf16* wbf    = (__bf16*)(ws + off); off = align256(off + (size_t)2 * NF * NF * 2);
    float*  pooled = (float*) (ws + off); off = align256(off + (size_t)NGRAPH * NF * 4);

    const int T = 256;
    const size_t NE_F = (size_t)N_EDGES * NF;   // 204.8M
    const size_t NN_F = (size_t)N_NODES * NF;   // 12.8M

    // weights -> bf16
    convert_w_bf16<<<(2 * NF * NF + T - 1) / T, T, 0, stream>>>(W1, W2, wbf);

    // degrees -> dinv
    zero_f32<<<(N_NODES + T - 1) / T, T, 0, stream>>>(dinv, N_NODES);
    deg_kernel<<<(N_EDGES + T - 1) / T, T, 0, stream>>>(dst, ew, dinv);
    finalize_dinv<<<(N_NODES + T - 1) / T, T, 0, stream>>>(dinv);

    // pooled accumulator
    zero_f32<<<(NGRAPH * NF + T - 1) / T, T, 0, stream>>>(pooled, NGRAPH * NF);

    const int LN_BLOCKS = (N_NODES * 32 + T - 1) / T;   // 8 nodes per 256-thr block

    // ---- layer 1 ----
    ln_to_bf16<<<LN_BLOCKS, T, 0, stream>>>(x, ln0g, ln0b, hbf, /*relu=*/0);
    gemm_bf16_wmma<<<N_NODES / 16, T, 0, stream>>>(hbf, wbf, hw);
    init_agg<<<(int)((NN_F + T - 1) / T), T, 0, stream>>>(hw, dinv, b1, agg);
    edge_scatter<<<(int)((NE_F + T - 1) / T), T, 0, stream>>>(src, dst, ew, dinv, hw, agg);

    // ---- layer 2 ----
    ln_to_bf16<<<LN_BLOCKS, T, 0, stream>>>(agg, ln1g, ln1b, hbf, /*relu=*/1);
    gemm_bf16_wmma<<<N_NODES / 16, T, 0, stream>>>(hbf, wbf + (size_t)NF * NF, hw);
    init_agg<<<(int)((NN_F + T - 1) / T), T, 0, stream>>>(hw, dinv, b2, agg);
    edge_scatter<<<(int)((NE_F + T - 1) / T), T, 0, stream>>>(src, dst, ew, dinv, hw, agg);

    // ---- head ----
    ln_relu_pool<<<LN_BLOCKS, T, 0, stream>>>(agg, ln2g, ln2b, batch, pooled);
    classifier<<<NGRAPH, 32, 0, stream>>>(pooled, Wc, bc, out);
}
